// Model_75814762709559
// MI455X (gfx1250) — matrix operand order repaired and checked by reading, not compiled, not run on hardware
//
#include <hip/hip_runtime.h>
#include <hip/hip_bf16.h>
#include <math.h>

typedef __attribute__((ext_vector_type(16))) __bf16 v16bf;
typedef __attribute__((ext_vector_type(8)))  __bf16 v8bf;
typedef __attribute__((ext_vector_type(8)))  float  v8f;

typedef int ai2 __attribute__((vector_size(2*sizeof(int))));
typedef int ai4 __attribute__((vector_size(4*sizeof(int))));

#define PI_F 3.14159265358979323846f

#define AS1 __attribute__((address_space(1)))
#define AS3 __attribute__((address_space(3)))

#if defined(__gfx1250__) && __has_builtin(__builtin_amdgcn_global_load_async_to_lds_b128) && \
    __has_builtin(__builtin_amdgcn_global_load_async_to_lds_b64) && \
    __has_builtin(__builtin_amdgcn_s_wait_asynccnt)
#define USE_ASYNC_LDS 1
#else
#define USE_ASYNC_LDS 0
#endif

#if USE_ASYNC_LDS
#define ASYNC_LDS_B128(g, l) \
  __builtin_amdgcn_global_load_async_to_lds_b128((AS1 ai4*)(void*)(g), (AS3 ai4*)(void*)(l), 0, 0)
#define ASYNC_LDS_B64(g, l) \
  __builtin_amdgcn_global_load_async_to_lds_b64((AS1 ai2*)(void*)(g), (AS3 ai2*)(void*)(l), 0, 0)
#endif

// ------------------------------------------------------------------ utils
__global__ void k_f2bf(const float* __restrict__ s, __bf16* __restrict__ d, size_t n){
  size_t i = (size_t)blockIdx.x*256u + threadIdx.x;
  if (i < n) d[i] = (__bf16)s[i];
}
__global__ void k_copy(const float* __restrict__ s, float* __restrict__ d, size_t n){
  size_t i = (size_t)blockIdx.x*256u + threadIdx.x;
  if (i < n) d[i] = s[i];
}
__global__ void k_zero(float* __restrict__ d, size_t n){
  size_t i = (size_t)blockIdx.x*256u + threadIdx.x;
  if (i < n) d[i] = 0.0f;
}

// ------------------------------------------------------------------ LDS FFT (radix-2, in place)
__device__ inline void fft_lds(float2* s, int n, int logn, float dir, int tid, int nt){
  for (int i = tid; i < n; i += nt){
    int j = (int)(__brev((unsigned)i) >> (32 - logn));
    if (j > i){ float2 t = s[i]; s[i] = s[j]; s[j] = t; }
  }
  __syncthreads();
  for (int st = 1; st <= logn; ++st){
    int m = 1 << st, hm = m >> 1;
    for (int b = tid; b < (n >> 1); b += nt){
      int pos = b & (hm - 1);
      int i1  = ((b >> (st - 1)) << st) + pos;
      int i2  = i1 + hm;
      float ang = dir * 2.0f * PI_F * (float)pos / (float)m;
      float sw, cw; __sincosf(ang, &sw, &cw);
      float2 u = s[i1], v = s[i2];
      float tx = v.x*cw - v.y*sw, ty = v.x*sw + v.y*cw;
      s[i1] = make_float2(u.x + tx, u.y + ty);
      s[i2] = make_float2(u.x - tx, u.y - ty);
    }
    __syncthreads();
  }
}

// ------------------------------------------------------------------ spectrogram (2048-pt FFT per frame)
__global__ void k_spectrogram(const float* __restrict__ xin, size_t rowstride,
                              float* __restrict__ spec, int subtract){
  __shared__ float2 s[2048];
  int f = blockIdx.x, b = blockIdx.y, tid = threadIdx.x;
  for (int w = tid; w < 2048; w += 256){
    int smp = f*256 + w;
    float v   = (smp < 32768) ? xin[(size_t)b*rowstride + smp] : 0.0f;
    float win = 0.5f - 0.5f*__cosf(2.0f*PI_F*(float)w/2048.0f);
    s[w] = make_float2(v*win, 0.0f);
  }
  __syncthreads();
  fft_lds(s, 2048, 11, -1.0f, tid, 256);
  for (int k = tid; k < 1024; k += 256){
    float m = sqrtf(s[k].x*s[k].x + s[k].y*s[k].y);
    size_t o = ((size_t)b*1024 + k)*128 + f;
    if (subtract) spec[o] -= m; else spec[o] = m;
  }
}

// ------------------------------------------------------------------ four-step 65536-pt FFT (256x256)
__global__ void k_fft_cols(const float2* __restrict__ in, float2* __restrict__ tmp, float dir){
  __shared__ __align__(16) float2 s[256];
  int n2 = blockIdx.x, r = blockIdx.y, tid = threadIdx.x;
  const float2* src = in  + (size_t)r*65536;
  float2*       dst = tmp + (size_t)r*65536;
#if USE_ASYNC_LDS
  for (int n1 = tid; n1 < 256; n1 += 128)
    ASYNC_LDS_B64(src + (size_t)n1*256 + n2, s + n1);
  __builtin_amdgcn_s_wait_asynccnt(0);
#else
  for (int n1 = tid; n1 < 256; n1 += 128) s[n1] = src[(size_t)n1*256 + n2];
#endif
  __syncthreads();
  fft_lds(s, 256, 8, dir, tid, 128);
  for (int k1 = tid; k1 < 256; k1 += 128){
    float ang = dir * 2.0f * PI_F * (float)(n2*k1) / 65536.0f;
    float sw, cw; __sincosf(ang, &sw, &cw);
    float2 v = s[k1];
    dst[(size_t)k1*256 + n2] = make_float2(v.x*cw - v.y*sw, v.x*sw + v.y*cw);
  }
}
__global__ void k_fft_rows(const float2* __restrict__ tmp, float2* __restrict__ out,
                           float dir, float scale){
  __shared__ __align__(16) float2 s[256];
  int k1 = blockIdx.x, r = blockIdx.y, tid = threadIdx.x;
  const float2* src = tmp + (size_t)r*65536 + (size_t)k1*256;
  float2*       dst = out + (size_t)r*65536;
#if USE_ASYNC_LDS
  ASYNC_LDS_B128(src + tid*2, s + tid*2);
  __builtin_amdgcn_s_wait_asynccnt(0);
#else
  for (int n2 = tid; n2 < 256; n2 += 128) s[n2] = src[n2];
#endif
  __syncthreads();
  fft_lds(s, 256, 8, dir, tid, 128);
  for (int k2 = tid; k2 < 256; k2 += 128)
    dst[(size_t)k2*256 + k1] = make_float2(s[k2].x*scale, s[k2].y*scale);
}
__global__ void k_cmul(float2* __restrict__ a, const float2* __restrict__ b){
  size_t i = ((size_t)blockIdx.y*65536) + (size_t)blockIdx.x*256 + threadIdx.x;
  float2 u = a[i], v = b[i];
  a[i] = make_float2(u.x*v.x - u.y*v.y, u.x*v.y + u.y*v.x);
}
__global__ void k_pack(const float* __restrict__ src, int srclen, size_t rowstride,
                       float2* __restrict__ dst){
  int i = blockIdx.x*256 + threadIdx.x; int r = blockIdx.y;
  float v = (i < srclen) ? src[(size_t)r*rowstride + i] : 0.0f;
  dst[(size_t)r*65536 + i] = make_float2(v, 0.0f);
}
__global__ void k_pack_kern(const float* __restrict__ rev, const float* __restrict__ dec,
                            float2* __restrict__ dst){
  int i = blockIdx.x*256 + threadIdx.x; int r = blockIdx.y;
  float v = (i < 32768) ? rev[i]*__expf(-dec[r]*(float)i/22050.0f) : 0.0f;
  dst[(size_t)r*65536 + i] = make_float2(v, 0.0f);
}

// ------------------------------------------------------------------ bf16 WMMA GEMM: C[M,N] = A[MxK] * Bt[NxK]^T
// A tile (shared by all 8 waves of the block) is double-buffered in LDS via
// async-to-LDS copies; B fragments stream from global (contiguous 32B/lane).
#define KC 512

__global__ void k_wmma_gemm_nt(const __bf16* __restrict__ A, const __bf16* __restrict__ Bt,
                               float* __restrict__ C, int M, int N, int K){
  __shared__ __align__(16) __bf16 As[2][16*KC];
  int wave = threadIdx.x >> 5;
  int lane = threadIdx.x & 31;
  int m0 = blockIdx.y * 16;
  int n0 = (blockIdx.x * 8 + wave) * 16;
  int r = lane & 15, half = lane >> 4;
  const __bf16* brow = Bt + (size_t)(n0 + r)*K + 16*half;   // B frag: K = 16*half+{0..15} contiguous
  const __bf16* al   = &As[0][0] + r*KC + 16*half;          // A frag from LDS
  int nch = K / KC;

  auto stage = [&](int ch, int bb){
    int kc = ch * KC;
    for (int q = 0; q < 4; ++q){                 // 16 rows * 512 bf16 = 1024 x 16B chunks
      int idx = q*256 + (int)threadIdx.x;
      int row = idx >> 6, c16 = idx & 63;
      const __bf16* g = A + (size_t)(m0 + row)*K + kc + c16*8;
      __bf16* l = &As[bb][row*KC + c16*8];
#if USE_ASYNC_LDS
      ASYNC_LDS_B128(g, l);
#else
      *(v8bf*)l = *(const v8bf*)g;
#endif
    }
  };

  v8f c = {};
  stage(0, 0);
  for (int ch = 0; ch < nch; ++ch){
    int bb = ch & 1;
    if (ch + 1 < nch){
      stage(ch + 1, bb ^ 1);
#if USE_ASYNC_LDS
      __builtin_amdgcn_s_wait_asynccnt(4);       // previous chunk's 4 per-wave copies done
#endif
    } else {
#if USE_ASYNC_LDS
      __builtin_amdgcn_s_wait_asynccnt(0);
#endif
    }
    __syncthreads();
    const __bf16* ab = al + bb*(16*KC);
    const __bf16* bk = brow + ch*KC;
#pragma unroll
    for (int kk = 0; kk < KC; kk += 32){
      union { v16bf v; v8bf h[2]; } a, b;
      a.h[0] = *(const v8bf*)(ab + kk);          // ds_load_b128
      a.h[1] = *(const v8bf*)(ab + kk + 8);
      b.v    = *(const v16bf*)(bk + kk);         // global b128 x2
      c = __builtin_amdgcn_wmma_f32_16x16x32_bf16(false, a.v, false, b.v, (short)0, c, false, false);
    }
    __syncthreads();                             // protect buffer about to be overwritten
  }
  if (m0 >= M || n0 >= N) return;
  float* crow = C + (size_t)(m0 + 8*half)*N + n0 + r;
#pragma unroll
  for (int v = 0; v < 8; ++v) crow[(size_t)v*N] = c[v];
}

// ------------------------------------------------------------------ encoder helpers
__global__ void k_prep_hcat(const float* __restrict__ h, __bf16* __restrict__ hcatT, int d){
  size_t id = (size_t)blockIdx.x*256 + threadIdx.x;   // 256 cols * 2048 k
  int col = (int)(id >> 11), k = (int)(id & 2047);
  int b = col >> 7, t = col & 127;
  int i = k >> 1, c = k & 1;
  int tt = t + c*d;
  float v = (tt < 128) ? h[((size_t)b*1024 + i)*128 + tt] : 0.0f;
  hcatT[id] = (__bf16)v;
}
__global__ void k_enc_epi(const float* __restrict__ yg, const float* __restrict__ bias,
                          float* __restrict__ h, float* __restrict__ acc){
  int o = blockIdx.x, col = threadIdx.x;
  float y = yg[(size_t)o*256 + col] + bias[o];
  y = (y > 0.0f) ? y : 0.2f*y;
  int b = col >> 7, t = col & 127;
  size_t hi = ((size_t)b*1024 + o)*128 + t;
  h[hi] += y; acc[hi] += y;
}
__global__ void k_ev_attn(const float* __restrict__ acc, const float* __restrict__ wv,
                          const float* __restrict__ bv, const float* __restrict__ wsw,
                          const float* __restrict__ bs, float* __restrict__ ev,
                          float* __restrict__ attn){
  __shared__ float red[256];
  int bt = blockIdx.x, b = bt >> 7, t = bt & 127, tid = threadIdx.x;
  float part[17];
#pragma unroll
  for (int j = 0; j < 17; ++j) part[j] = 0.0f;
  for (int i = tid; i < 1024; i += 256){
    float e = acc[((size_t)b*1024 + i)*128 + t];
#pragma unroll
    for (int c = 0; c < 16; ++c) part[c] += wv[(size_t)c*1024 + i]*e;
    part[16] += wsw[i]*e;
  }
#pragma unroll
  for (int j = 0; j < 17; ++j){
    red[tid] = part[j]; __syncthreads();
    for (int s2 = 128; s2 > 0; s2 >>= 1){ if (tid < s2) red[tid] += red[tid+s2]; __syncthreads(); }
    if (tid == 0){
      if (j < 16) ev[(size_t)bt*16 + j] = red[0] + bv[j];
      else { float a = red[0] + bs[0]; attn[bt] = (a > 0.0f) ? a : 0.0f; }
    }
    __syncthreads();
  }
}
__global__ void k_argmax(const float* __restrict__ attn, const float* __restrict__ ev, int e,
                         int* __restrict__ idx_all, float* __restrict__ val_all,
                         float* __restrict__ o_sched, float* __restrict__ o_vecs){
  __shared__ float vmax[128]; __shared__ int vidx[128];
  int b = blockIdx.x, t = threadIdx.x;
  vmax[t] = attn[b*128 + t]; vidx[t] = t;
  __syncthreads();
  for (int s2 = 64; s2 > 0; s2 >>= 1){
    if (t < s2 && vmax[t+s2] > vmax[t]){ vmax[t] = vmax[t+s2]; vidx[t] = vidx[t+s2]; }
    __syncthreads();
  }
  int im = vidx[0]; float vm = vmax[0];
  if (t == 0){ idx_all[b*16 + e] = im; val_all[b*16 + e] = vm; }
  o_sched[((size_t)(b*16 + e))*128 + t] = (t == im) ? vm : 0.0f;
  if (t < 16) o_vecs[((size_t)(b*16 + e))*16 + t] = ev[((size_t)b*128 + im)*16 + t];
}

// ------------------------------------------------------------------ generate helpers
__global__ void k_gather(const float* __restrict__ o_vecs, const float* __restrict__ val_all,
                         const int* __restrict__ idx_all, int n, int e0,
                         float* __restrict__ vecbuf, float* __restrict__ amps,
                         int* __restrict__ shift, int R){
  int r = threadIdx.x; if (r >= R) return;
  int b = r / n, e = e0 + (r % n);
  amps[r]  = val_all[b*16 + e];
  shift[r] = idx_all[b*16 + e] * 256;
  for (int c = 0; c < 16; ++c) vecbuf[r*16 + c] = o_vecs[((size_t)(b*16 + e))*16 + c];
}
__global__ void k_emb(const float* __restrict__ vecbuf, const float* __restrict__ Wc,
                      const float* __restrict__ bc, float* __restrict__ emb){
  int r = blockIdx.x, o = threadIdx.x;
  float a = bc[o];
  for (int c = 0; c < 16; ++c) a += vecbuf[r*16 + c]*Wc[c*256 + o];
  emb[(size_t)r*256 + o] = a;
}
__global__ void k_env(const float* __restrict__ emb, const float* __restrict__ We,
                      const float* __restrict__ be, float* __restrict__ env){
  int r = blockIdx.x, f = threadIdx.x;
  float a = be[f];
  for (int o = 0; o < 256; ++o) a += emb[(size_t)r*256 + o]*We[o*128 + f];
  env[r*128 + f] = 1.0f/(1.0f + __expf(-a));
}
__global__ void k_logits(const float* __restrict__ emb, const float* __restrict__ Ws,
                         const float* __restrict__ bsel, const float* __restrict__ abias,
                         float* __restrict__ sel){
  int a = blockIdx.x*256 + threadIdx.x, r = blockIdx.y;
  float s = bsel[a] + abias[a];
  for (int o = 0; o < 256; ++o) s += emb[(size_t)r*256 + o]*Ws[(size_t)o*4096 + a];
  sel[(size_t)r*4096 + a] = s;
}
__global__ void k_softmax(float* __restrict__ sel){
  __shared__ float red[256];
  int r = blockIdx.x, tid = threadIdx.x;
  float mx = -1e30f;
  for (int a = tid; a < 4096; a += 256) mx = fmaxf(mx, sel[(size_t)r*4096 + a]);
  red[tid] = mx; __syncthreads();
  for (int s2 = 128; s2 > 0; s2 >>= 1){ if (tid < s2) red[tid] = fmaxf(red[tid], red[tid+s2]); __syncthreads(); }
  mx = red[0]; __syncthreads();
  float se = 0.0f;
  for (int a = tid; a < 4096; a += 256){
    float e = __expf(sel[(size_t)r*4096 + a] - mx);
    sel[(size_t)r*4096 + a] = e; se += e;
  }
  red[tid] = se; __syncthreads();
  for (int s2 = 128; s2 > 0; s2 >>= 1){ if (tid < s2) red[tid] += red[tid+s2]; __syncthreads(); }
  float inv = 1.0f/red[0];
  for (int a = tid; a < 4096; a += 256) sel[(size_t)r*4096 + a] *= inv;
}
__global__ void k_imp(const float* __restrict__ env, const float* __restrict__ inoise,
                      float* __restrict__ dst){
  __shared__ float red[256];
  int r = blockIdx.x, tid = threadIdx.x;
  float ssq = 0.0f;
  for (int j = tid; j < 16384; j += 256){
    float v = env[r*128 + (j >> 7)]*inoise[j];
    dst[(size_t)r*16384 + j] = v; ssq += v*v;
  }
  red[tid] = ssq; __syncthreads();
  for (int s2 = 128; s2 > 0; s2 >>= 1){ if (tid < s2) red[tid] += red[tid+s2]; __syncthreads(); }
  float sc = 1.0f/(sqrtf(red[0]) + 1e-8f);
  for (int j = tid; j < 16384; j += 256) dst[(size_t)r*16384 + j] *= sc;
}
template<int R>
__global__ void k_resn(const float* __restrict__ sel, const float* __restrict__ waves,
                       float* __restrict__ resn){
  int t = blockIdx.x*256 + threadIdx.x;
  float acc[R] = {0.0f};
  for (int a = 0; a < 4096; ++a){
    if (a + 16 < 4096) __builtin_prefetch(waves + (size_t)(a + 16)*32768 + t, 0, 0);
    float w = waves[(size_t)a*32768 + t];
#pragma unroll
    for (int r = 0; r < R; ++r) acc[r] += sel[(size_t)r*4096 + a]*w;
  }
#pragma unroll
  for (int r = 0; r < R; ++r) resn[(size_t)r*32768 + t] = acc[r];
}
__global__ void k_post_mixed(const float2* __restrict__ X, const float* __restrict__ amps,
                             float* __restrict__ dst){
  __shared__ float red[256];
  int r = blockIdx.x, tid = threadIdx.x;
  float ssq = 0.0f;
  for (int t = tid; t < 32768; t += 256){ float v = X[(size_t)r*65536 + t].x; ssq += v*v; }
  red[tid] = ssq; __syncthreads();
  for (int s2 = 128; s2 > 0; s2 >>= 1){ if (tid < s2) red[tid] += red[tid+s2]; __syncthreads(); }
  float sc = amps[r]/(sqrtf(red[0]) + 1e-8f);
  for (int t = tid; t < 32768; t += 256)
    dst[(size_t)r*32768 + t] = X[(size_t)r*65536 + t].x * sc;
}
__global__ void k_dry(const float* __restrict__ mixed, const float* __restrict__ amps,
                      const int* __restrict__ shift, float* __restrict__ dry){
  int r = blockIdx.y, t = blockIdx.x*256 + threadIdx.x;
  int s = shift[r];
  dry[(size_t)r*32768 + t] = (t >= s) ? mixed[(size_t)r*32768 + (t - s)]*amps[r] : 0.0f;
}
__global__ void k_mixdec(const float* __restrict__ vecbuf, const float* __restrict__ lg,
                         const float* __restrict__ lb, const float* __restrict__ Wm,
                         const float* __restrict__ bm, const float* __restrict__ Wd,
                         const float* __restrict__ bd, float* __restrict__ mix,
                         float* __restrict__ dec, int R){
  int r = threadIdx.x; if (r >= R) return;
  float v[16], ssq = 0.0f;
  for (int c = 0; c < 16; ++c){ v[c] = vecbuf[r*16 + c]; ssq += v[c]*v[c]; }
  float inv = 1.0f/(sqrtf(ssq) + 1e-8f);
  float mean = 0.0f;
  for (int c = 0; c < 16; ++c){ v[c] *= inv; mean += v[c]; }
  mean *= (1.0f/16.0f);
  float var = 0.0f;
  for (int c = 0; c < 16; ++c){ float d = v[c] - mean; var += d*d; }
  var *= (1.0f/16.0f);
  float isd = 1.0f/(sqrtf(var) + 1e-5f);
  float sm = 0.0f, sd2 = 0.0f;
  for (int c = 0; c < 16; ++c){
    float h = (v[c] - mean)*isd*lg[c] + lb[c];
    sm += h*Wm[c]; sd2 += h*Wd[c];
  }
  mix[r] = 1.0f/(1.0f + __expf(-(sm + bm[0])));
  dec[r] = 2.0f + 10.0f/(1.0f + __expf(-(sd2 + bd[0])));
}
__global__ void k_final(const float2* __restrict__ X, const float* __restrict__ dry,
                        const float* __restrict__ mix, float* __restrict__ out, int e){
  int r = blockIdx.y, t = blockIdx.x*256 + threadIdx.x;   // in-loop: r == b, n == 1
  float wet = X[(size_t)r*65536 + t].x;
  float d   = dry[(size_t)r*32768 + t];
  float m   = mix[r];
  out[((size_t)(r*16 + e))*32768 + t] = m*wet + (1.0f - m)*d;
}

// ------------------------------------------------------------------ host
extern "C" void kernel_launch(void* const* d_in, const int* in_sizes, int n_in,
                              void* d_out, int out_size, void* d_ws, size_t ws_size,
                              hipStream_t stream){
  const float* x         = (const float*)d_in[0];
  const float* enc_w     = (const float*)d_in[1];
  const float* enc_b     = (const float*)d_in[2];
  const float* wv        = (const float*)d_in[3];
  const float* bv        = (const float*)d_in[4];
  const float* ws_w      = (const float*)d_in[5];
  const float* bs        = (const float*)d_in[6];
  const float* W_ctx     = (const float*)d_in[7];
  const float* b_ctx     = (const float*)d_in[8];
  const float* W_env     = (const float*)d_in[9];
  const float* b_env     = (const float*)d_in[10];
  const float* W_sel     = (const float*)d_in[11];
  const float* b_sel     = (const float*)d_in[12];
  const float* atom_bias = (const float*)d_in[13];
  const float* waves     = (const float*)d_in[14];
  const float* imp_noise = (const float*)d_in[15];
  const float* rev_noise = (const float*)d_in[16];
  const float* ln_g      = (const float*)d_in[17];
  const float* ln_b      = (const float*)d_in[18];
  const float* W_mix     = (const float*)d_in[19];
  const float* b_mix     = (const float*)d_in[20];
  const float* W_dec     = (const float*)d_in[21];
  const float* b_dec     = (const float*)d_in[22];

  float* out    = (float*)d_out;
  float* o_ch   = out;                    // (2,16,32768)
  float* o_vecs = out + 1048576;          // (2,16,16)
  float* o_imp  = out + 1049088;          // (2,16,16384)
  float* o_sch  = out + 1573376;          // (2,16,128)
  float* o_amps = out + 1577472;          // (2,16,1)
  float* o_mix  = out + 1577504;          // (2,16,32768)

  char* wp = (char*)d_ws;
  auto carve = [&](size_t bytes)->void*{ void* p = (void*)wp; wp += (bytes + 255) & ~(size_t)255; return p; };
  __bf16* enc_w_bf = (__bf16*)carve(8ull*1024*2048*2);
  float*  spec     = (float*) carve(262144ull*4);
  float*  hbuf     = (float*) carve(262144ull*4);
  float*  accb     = (float*) carve(262144ull*4);
  __bf16* hcatT    = (__bf16*)carve(524288ull*2);
  float*  ygemm    = (float*) carve(262144ull*4);
  float*  ev       = (float*) carve(2ull*128*16*4);
  float*  attn     = (float*) carve(256ull*4);
  int*    idx_all  = (int*)   carve(32ull*4);
  float*  val_all  = (float*) carve(32ull*4);
  float*  vecbuf   = (float*) carve(32ull*16*4);
  float*  ampsb    = (float*) carve(32ull*4);
  int*    shiftb   = (int*)   carve(32ull*4);
  float*  mixb     = (float*) carve(32ull*4);
  float*  decb     = (float*) carve(32ull*4);
  float*  emb      = (float*) carve(32ull*256*4);
  float*  envb     = (float*) carve(32ull*128*4);
  float*  selb     = (float*) carve(32ull*4096*4);
  float*  impb     = (float*) carve(32ull*16384*4);
  float*  resnb    = (float*) carve(32ull*32768*4);
  float*  mixedb   = (float*) carve(32ull*32768*4);
  float*  dryb     = (float*) carve(32ull*32768*4);
  float2* X1       = (float2*)carve(32ull*65536*8);
  float2* X2       = (float2*)carve(32ull*65536*8);
  float2* XT       = (float2*)carve(32ull*65536*8);

  auto fftf = [&](float2* X, int R){
    k_fft_cols<<<dim3(256,R),128,0,stream>>>(X, XT, -1.0f);
    k_fft_rows<<<dim3(256,R),128,0,stream>>>(XT, X, -1.0f, 1.0f);
  };
  auto ffti = [&](float2* X, int R){
    k_fft_cols<<<dim3(256,R),128,0,stream>>>(X, XT, 1.0f);
    k_fft_rows<<<dim3(256,R),128,0,stream>>>(XT, X, 1.0f, 1.0f/65536.0f);
  };

  const int DIL[8] = {1,2,4,8,16,32,64,1};

  k_f2bf<<<65536,256,0,stream>>>(enc_w, enc_w_bf, 16777216ull);
  k_spectrogram<<<dim3(128,2),256,0,stream>>>(x, 32768, spec, 0);

  for (int e = 0; e < 16; ++e){
    // ---- encode
    k_copy<<<1024,256,0,stream>>>(spec, hbuf, 262144ull);
    k_zero<<<1024,256,0,stream>>>(accb, 262144ull);
    for (int l = 0; l < 8; ++l){
      k_prep_hcat<<<2048,256,0,stream>>>(hbuf, hcatT, DIL[l]);
      k_wmma_gemm_nt<<<dim3(2,64),256,0,stream>>>(enc_w_bf + (size_t)l*1024*2048, hcatT,
                                                  ygemm, 1024, 256, 2048);
      k_enc_epi<<<1024,256,0,stream>>>(ygemm, enc_b + l*1024, hbuf, accb);
    }
    k_ev_attn<<<256,256,0,stream>>>(accb, wv, bv, ws_w, bs, ev, attn);
    k_argmax<<<2,128,0,stream>>>(attn, ev, e, idx_all, val_all, o_sch, o_vecs);

    // ---- generate (n = 1, R = 2)
    k_gather<<<1,32,0,stream>>>(o_vecs, val_all, idx_all, 1, e, vecbuf, ampsb, shiftb, 2);
    k_emb<<<2,256,0,stream>>>(vecbuf, W_ctx, b_ctx, emb);
    k_env<<<2,128,0,stream>>>(emb, W_env, b_env, envb);
    k_logits<<<dim3(16,2),256,0,stream>>>(emb, W_sel, b_sel, atom_bias, selb);
    k_softmax<<<2,256,0,stream>>>(selb);
    k_imp<<<2,256,0,stream>>>(envb, imp_noise, impb);
    k_resn<2><<<128,256,0,stream>>>(selb, waves, resnb);
    // mixed = unit_norm(conv(imp, resn)) * amps
    k_pack<<<dim3(256,2),256,0,stream>>>(impb, 16384, 16384, X1);
    k_pack<<<dim3(256,2),256,0,stream>>>(resnb, 32768, 32768, X2);
    fftf(X1,2); fftf(X2,2);
    k_cmul<<<dim3(256,2),256,0,stream>>>(X1, X2);
    ffti(X1,2);
    k_post_mixed<<<2,256,0,stream>>>(X1, ampsb, mixedb);
    // dry = shifted/scaled mixed (sched is one-hot)
    k_dry<<<dim3(128,2),256,0,stream>>>(mixedb, ampsb, shiftb, dryb);
    // reverb
    k_mixdec<<<1,32,0,stream>>>(vecbuf, ln_g, ln_b, W_mix, b_mix, W_dec, b_dec, mixb, decb, 2);
    k_pack<<<dim3(256,2),256,0,stream>>>(dryb, 32768, 32768, X1);
    k_pack_kern<<<dim3(256,2),256,0,stream>>>(rev_noise, decb, X2);
    fftf(X1,2); fftf(X2,2);
    k_cmul<<<dim3(256,2),256,0,stream>>>(X1, X2);
    ffti(X1,2);
    k_final<<<dim3(128,2),256,0,stream>>>(X1, dryb, mixb, o_ch, e);
    // spec -= spectrogram(ch)
    k_spectrogram<<<dim3(128,2),256,0,stream>>>(o_ch + (size_t)e*32768, 16ull*32768, spec, 1);
  }

  // ---- final generate over all 16 events (only imp/amps/mixed are returned)
  k_gather<<<1,32,0,stream>>>(o_vecs, val_all, idx_all, 16, 0, vecbuf, ampsb, shiftb, 32);
  k_emb<<<32,256,0,stream>>>(vecbuf, W_ctx, b_ctx, emb);
  k_env<<<32,128,0,stream>>>(emb, W_env, b_env, envb);
  k_logits<<<dim3(16,32),256,0,stream>>>(emb, W_sel, b_sel, atom_bias, selb);
  k_softmax<<<32,256,0,stream>>>(selb);
  k_imp<<<32,256,0,stream>>>(envb, imp_noise, o_imp);
  k_resn<32><<<128,256,0,stream>>>(selb, waves, resnb);
  k_pack<<<dim3(256,32),256,0,stream>>>(o_imp, 16384, 16384, X1);
  k_pack<<<dim3(256,32),256,0,stream>>>(resnb, 32768, 32768, X2);
  fftf(X1,32); fftf(X2,32);
  k_cmul<<<dim3(256,32),256,0,stream>>>(X1, X2);
  ffti(X1,32);
  k_post_mixed<<<32,256,0,stream>>>(X1, ampsb, o_mix);
  k_copy<<<1,256,0,stream>>>(ampsb, o_amps, 32ull);
}
